// JointNet_67336497266979
// MI455X (gfx1250) — matrix operand order repaired and checked by reading, not compiled, not run on hardware
//
#include <hip/hip_runtime.h>

// Problem constants
#define B_ 4
#define T_ 512
#define U_ 128
#define D_ 256
#define H_ 512
#define V_ 512

#define MT 64     // M rows per workgroup tile ((t,u) pairs)
#define NT 256    // N (V) columns per workgroup
#define KT 32     // K step per WMMA (bf16 16x16x32)
#define NK (H_ / KT)
#define LDA 40    // padded LDS row stride in bf16 units (32 data + 8 pad)

typedef __attribute__((ext_vector_type(16))) __bf16    v16bf;
typedef __attribute__((ext_vector_type(8)))  float     v8f;
typedef __attribute__((ext_vector_type(4)))  float     v4f;
typedef __attribute__((ext_vector_type(4)))  unsigned int v4u;
typedef __attribute__((ext_vector_type(4)))  int       v4i;

union Frag16 { v4u q[2]; v16bf v; };

#if __has_builtin(__builtin_amdgcn_global_load_async_to_lds_b128)
#define HAVE_ASYNC 1
typedef __attribute__((address_space(1))) v4i gv4i_t;   // global v4i
typedef __attribute__((address_space(3))) v4i lv4i_t;   // LDS v4i
#else
#define HAVE_ASYNC 0
#endif

__device__ __forceinline__ void wait_async_all() {
#if HAVE_ASYNC
#if __has_builtin(__builtin_amdgcn_s_wait_asynccnt)
  __builtin_amdgcn_s_wait_asynccnt(0);
#else
  asm volatile("s_wait_asynccnt 0x0" ::: "memory");
#endif
#endif
}

__device__ __forceinline__ unsigned short f2bf(float f) {
  unsigned int u = __float_as_uint(f);
  u += 0x7FFFu + ((u >> 16) & 1u);   // round-to-nearest-even
  return (unsigned short)(u >> 16);
}

// gelu(tanh approx): x * sigmoid(1.5957691*x + 0.0713548*x^3)
__device__ __forceinline__ float gelu_tanh(float s) {
  float s2 = s * s;
  float z  = s * __builtin_fmaf(0.0713548162726f, s2, 1.5957691216057308f);
  float e  = __builtin_amdgcn_exp2f(z * -1.4426950408889634f); // exp(-z)
  return s * __builtin_amdgcn_rcpf(1.0f + e);
}

// dst[row*H + h] = src @ W (+ optional b1). W is a (D_,H_) row-major slice of W1.
__global__ void proj_kernel(const float* __restrict__ src, const float* __restrict__ W,
                            const float* __restrict__ bias, int useBias,
                            float* __restrict__ dst, int rows) {
  int idx = blockIdx.x * blockDim.x + threadIdx.x;
  int h   = idx & (H_ - 1);
  int row = idx >> 9;                  // /H_
  if (row >= rows) return;
  float sum = useBias ? bias[h] : 0.0f;
  const float* s = src + (size_t)row * D_;
  #pragma unroll 4
  for (int d = 0; d < D_; ++d)
    sum = __builtin_fmaf(s[d], W[(size_t)d * H_ + h], sum);
  dst[idx] = sum;
}

// WfcT[v*H + h] = bf16(Wfc[h*V + v])  (transpose so K runs contiguously per column)
__global__ void cvt_wfc_kernel(const float* __restrict__ Wfc,
                               unsigned short* __restrict__ WfcT) {
  int idx = blockIdx.x * blockDim.x + threadIdx.x;  // = v*H_ + h
  int h = idx & (H_ - 1);
  int v = idx >> 9;
  WfcT[idx] = f2bf(Wfc[(size_t)h * V_ + v]);
}

// Fused gelu(pe+pd) @ Wfc, bf16 WMMA, f32 accumulate, double-buffered LDS,
// async global->LDS staging of the B (WfcT) tiles.
__global__ void __launch_bounds__(256)
joint_kernel(const float* __restrict__ pe, const float* __restrict__ pd,
             const unsigned short* __restrict__ WfcT, float* __restrict__ out) {
  __shared__ __align__(16) unsigned short sA[2][MT * LDA];   // 2 x 5 KB
  __shared__ __align__(16) unsigned short sB[2][NT * LDA];   // 2 x 20 KB

  const int tid = threadIdx.x;
  const int b   = blockIdx.z;
  const int m0  = blockIdx.y * MT;   // M tile base ((t,u) flat index)
  const int n0  = blockIdx.x * NT;   // V column base

  // ---- A producer mapping: thread -> (row, 8-wide K chunk) ----
  const int ar = tid >> 2;                 // 0..63
  const int ac = (tid & 3) * 8;            // 0,8,16,24
  const int am = m0 + ar;
  const float* peRow = pe + ((size_t)b * T_ + (am >> 7)) * H_ + ac;
  const float* pdRow = pd + ((size_t)b * U_ + (am & (U_ - 1))) * H_ + ac;

  // ---- B producer mapping: one V-column (64 B of K) per thread ----
  const unsigned short* wCol = WfcT + (size_t)(n0 + tid) * H_;

  // ---- wave layout: 8 waves as 2(M) x 4(N), each owns 32M x 64N ----
  const int wave = tid >> 5;
  const int lane = tid & 31;
  const int half = lane >> 4;
  const int l16  = lane & 15;
  const int wm   = wave >> 2;              // 0..1
  const int wn   = wave & 3;               // 0..3

  auto produceA = [&](int buf, int K0) {
    v4f p0 = *(const v4f*)(peRow + K0);
    v4f p1 = *(const v4f*)(peRow + K0 + 4);
    v4f q0 = *(const v4f*)(pdRow + K0);
    v4f q1 = *(const v4f*)(pdRow + K0 + 4);
    float hv[8];
    #pragma unroll
    for (int i = 0; i < 4; ++i) hv[i]     = gelu_tanh(p0[i] + q0[i]);
    #pragma unroll
    for (int i = 0; i < 4; ++i) hv[4 + i] = gelu_tanh(p1[i] + q1[i]);
    v4u packed;
    #pragma unroll
    for (int i = 0; i < 4; ++i)
      packed[i] = (unsigned int)f2bf(hv[2 * i]) |
                  ((unsigned int)f2bf(hv[2 * i + 1]) << 16);
    *(v4u*)&sA[buf][ar * LDA + ac] = packed;
  };

  auto stageB = [&](int buf, int K0) {
#if HAVE_ASYNC
    gv4i_t* gp = (gv4i_t*)(wCol + K0);
    lv4i_t* lp = (lv4i_t*)&sB[buf][tid * LDA];
    __builtin_amdgcn_global_load_async_to_lds_b128(gp, lp, 0, 0);
    __builtin_amdgcn_global_load_async_to_lds_b128(gp, lp, 16, 0);
    __builtin_amdgcn_global_load_async_to_lds_b128(gp, lp, 32, 0);
    __builtin_amdgcn_global_load_async_to_lds_b128(gp, lp, 48, 0);
#else
    const v4u* gs = (const v4u*)(wCol + K0);
    v4u* ld = (v4u*)&sB[buf][tid * LDA];
    #pragma unroll
    for (int i = 0; i < 4; ++i) ld[i] = gs[i];
#endif
  };

  v8f acc[2][4] = {};

  // prologue: stage K-step 0 into buffer 0
  stageB(0, 0);
  produceA(0, 0);
  wait_async_all();
  __syncthreads();

  for (int ks = 0; ks < NK; ++ks) {
    const int cur = ks & 1;
    if (ks + 1 < NK) {               // stage next K-step into the other buffer
      stageB(cur ^ 1, (ks + 1) * KT);
      produceA(cur ^ 1, (ks + 1) * KT);
    }
    // ---- consume current buffers: 2 A frags, 4 B frags, 8 WMMAs ----
    Frag16 a[2];
    #pragma unroll
    for (int mr = 0; mr < 2; ++mr) {
      const unsigned short* ap = &sA[cur][(wm * 32 + mr * 16 + l16) * LDA + 8 * half];
      a[mr].q[0] = *(const v4u*)(ap);        // K = 8*half .. +7
      a[mr].q[1] = *(const v4u*)(ap + 16);   // K = 16+8*half .. +7
    }
    #pragma unroll
    for (int j = 0; j < 4; ++j) {
      const unsigned short* bp = &sB[cur][(wn * 64 + j * 16 + l16) * LDA + half * 8];
      Frag16 bb;
      bb.q[0] = *(const v4u*)(bp);           // K = 8*half .. +7
      bb.q[1] = *(const v4u*)(bp + 16);      // K = 16+8*half .. +7
      acc[0][j] = __builtin_amdgcn_wmma_f32_16x16x32_bf16(
          false, a[0].v, false, bb.v, (short)0, acc[0][j], false, false);
      acc[1][j] = __builtin_amdgcn_wmma_f32_16x16x32_bf16(
          false, a[1].v, false, bb.v, (short)0, acc[1][j], false, false);
    }
    wait_async_all();
    __syncthreads();
  }

  // ---- epilogue: C/D layout -> out[(b*T*U + m)*V + col] ----
  #pragma unroll
  for (int mr = 0; mr < 2; ++mr) {
    const size_t rowBase =
        ((size_t)b * (T_ * U_) + (size_t)(m0 + wm * 32 + mr * 16 + 8 * half)) * V_;
    #pragma unroll
    for (int j = 0; j < 4; ++j) {
      const int col = n0 + wn * 64 + j * 16 + l16;
      #pragma unroll
      for (int r = 0; r < 8; ++r)
        out[rowBase + (size_t)r * V_ + col] = acc[mr][j][r];
    }
  }
}

extern "C" void kernel_launch(void* const* d_in, const int* in_sizes, int n_in,
                              void* d_out, int out_size, void* d_ws, size_t ws_size,
                              hipStream_t stream) {
  const float* enc = (const float*)d_in[0];
  const float* dec = (const float*)d_in[1];
  const float* W1  = (const float*)d_in[2];
  const float* b1  = (const float*)d_in[3];
  const float* Wfc = (const float*)d_in[4];
  float* out = (float*)d_out;

  char* ws = (char*)d_ws;
  unsigned short* WfcT = (unsigned short*)ws;                       // 512 KB
  float* pe = (float*)(ws + 512 * 1024);                            // 4 MB
  float* pd = (float*)(ws + 512 * 1024 + 4 * 1024 * 1024);          // 1 MB

  cvt_wfc_kernel<<<(H_ * V_) / 256, 256, 0, stream>>>(Wfc, WfcT);
  proj_kernel<<<(B_ * T_ * H_) / 256, 256, 0, stream>>>(enc, W1, b1, 1, pe, B_ * T_);
  proj_kernel<<<(B_ * U_ * H_) / 256, 256, 0, stream>>>(dec, W1 + D_ * H_, b1, 0, pd, B_ * U_);

  dim3 grid(V_ / NT, (T_ * U_) / MT, B_);
  joint_kernel<<<grid, 256, 0, stream>>>(pe, pd, WfcT, out);
}
